// LinearTransformer_18391049961862
// MI455X (gfx1250) — matrix operand order repaired and checked by reading, not compiled, not run on hardware
//
#include <hip/hip_runtime.h>
#include <hip/hip_bf16.h>

// ---------------------------------------------------------------------------
// Linear-attention transformer forward for MI455X (gfx1250), bf16 WMMA.
//   - all GEMMs: v_wmma_f32_16x16x32_bf16, 64x64 tile per wave, 8 waves/block,
//     software-pipelined (double-buffered) k-loop, sched_group_barrier
//     enforced load/WMMA interleave
//   - attention: chunked causal linear attention, WMMA for all matmuls,
//     LDS for layout conversion + running (d x e) state.
// ---------------------------------------------------------------------------

typedef __attribute__((ext_vector_type(16))) __bf16 v16bf;
typedef __attribute__((ext_vector_type(8)))  float  v8f;

// sched_group_barrier masks
#define SG_WMMA   0x008
#define SG_VMREAD 0x020
#define SG_DSREAD 0x100
#define SGB(mask, size) __builtin_amdgcn_sched_group_barrier((mask), (size), 0)

union FragBF {
    v16bf v;
    uint4 u[2];
    __bf16 h[16];
};

__device__ __forceinline__ float  bf2f(__bf16 x) { return (float)x; }
__device__ __forceinline__ __bf16 f2bf(float x)  { return (__bf16)x; }

__device__ __forceinline__ v8f wmma_bf16(v16bf a, v16bf b, v8f c) {
    return __builtin_amdgcn_wmma_f32_16x16x32_bf16(
        false, a, false, b, (short)0, c, false, false);
}

// model constants
#define MB_   4
#define SEQ_  2048
#define DMODEL 512
#define NHEAD 8
#define DHEAD 64
#define CHK   128
#define NCHK  (SEQ_ / CHK)
#define MTOK  (MB_ * SEQ_)   // 8192 rows
#define EMBCAT 1216

// ---------------------------------------------------------------------------
// Embedding gather + scale -> bf16 [MTOK, 1216]
// ---------------------------------------------------------------------------
__global__ __launch_bounds__(256) void embed_k(
    const int* __restrict__ x,
    const float* __restrict__ e0, const float* __restrict__ e1,
    const float* __restrict__ e2, const float* __restrict__ e3,
    const float* __restrict__ e4, const float* __restrict__ e5,
    __bf16* __restrict__ out)
{
    int i = blockIdx.x * 256 + threadIdx.x;
    const int total = MTOK * EMBCAT;
    if (i >= total) return;
    int m = i / EMBCAT;
    int c = i - m * EMBCAT;
    int ti, off, dim;
    const float* tab;
    if      (c <  128) { ti = 0; off = 0;    dim = 128; tab = e0; }
    else if (c <  384) { ti = 1; off = 128;  dim = 256; tab = e1; }
    else if (c <  448) { ti = 2; off = 384;  dim = 64;  tab = e2; }
    else if (c <  960) { ti = 3; off = 448;  dim = 512; tab = e3; }
    else if (c < 1088) { ti = 4; off = 960;  dim = 128; tab = e4; }
    else               { ti = 5; off = 1088; dim = 128; tab = e5; }
    int tok = x[m * 6 + ti];
    float v = tab[(size_t)tok * dim + (c - off)] * sqrtf((float)dim);
    out[(size_t)m * EMBCAT + c] = f2bf(v);
}

// ---------------------------------------------------------------------------
// Weight transpose + convert: W fp32 [K,N] -> WT bf16 [N,K].
// LDS-tiled 32x32 (padded 33) so both the read and the write are coalesced.
// Requires K % 32 == 0 and N % 32 == 0 (true for every weight here).
// ---------------------------------------------------------------------------
__global__ __launch_bounds__(256) void transposeW(
    const float* __restrict__ W, __bf16* __restrict__ WT, int K, int N)
{
    __shared__ float t[32][33];
    const int tilesN = N >> 5;
    const int tk = (blockIdx.x / tilesN) << 5;
    const int tn = (blockIdx.x % tilesN) << 5;
    const int r = threadIdx.x >> 5;    // 0..7
    const int c = threadIdx.x & 31;
#pragma unroll
    for (int i = 0; i < 4; ++i)
        t[r + i * 8][c] = W[(size_t)(tk + r + i * 8) * N + tn + c];
    __syncthreads();
#pragma unroll
    for (int i = 0; i < 4; ++i)
        WT[(size_t)(tn + r + i * 8) * K + tk + c] = f2bf(t[c][r + i * 8]);
}

// ---------------------------------------------------------------------------
// WMMA GEMM: out = A[M,K](bf16) @ WT[N,K]^T (bf16) + bias (+epilogue)
// FLAGS: 1=ELU1 (elu(x)+1), 2=GELU(exact), 4=positional enc, 8=residual add
// K must be a multiple of 64 (512 / 1216 / 2048 here).
// Software-pipelined: fragments double-buffered across 32-deep k-steps;
// sched_group_barrier pins (2 loads, 2 WMMA) interleave so next-stage loads
// stay in flight under current-stage WMMAs.
// ---------------------------------------------------------------------------
template<int FLAGS>
__global__ __launch_bounds__(256) void gemm_k(
    const __bf16* __restrict__ A, const __bf16* __restrict__ WT,
    const float* __restrict__ bias, const float* __restrict__ resid,
    float* __restrict__ outF, __bf16* __restrict__ outB,
    int M, int N, int K)
{
    constexpr bool ELU1 = (FLAGS & 1) != 0;
    constexpr bool GELU = (FLAGS & 2) != 0;
    constexpr bool PENC = (FLAGS & 4) != 0;
    constexpr bool RESD = (FLAGS & 8) != 0;

    const int lane   = threadIdx.x & 31;
    const int wave   = threadIdx.x >> 5;
    const int l16    = lane & 15;
    const int half16 = lane >> 4;

    const int tilesN = (N + 63) >> 6;
    const int tilesM = M >> 6;
    int t = blockIdx.x * 8 + wave;
    if (t >= tilesM * tilesN) return;   // wave-uniform
    const int tm = (t / tilesN) << 6;
    const int tn = (t % tilesN) << 6;

    v8f acc[4][4];
#pragma unroll
    for (int i = 0; i < 4; ++i)
#pragma unroll
        for (int j = 0; j < 4; ++j) acc[i][j] = (v8f){0,0,0,0,0,0,0,0};

    const __bf16* Arow[4];
#pragma unroll
    for (int i = 0; i < 4; ++i)
        Arow[i] = A + (size_t)(tm + i * 16 + l16) * K + half16 * 16;

    const __bf16* Brow[4];
    bool nok[4];
#pragma unroll
    for (int j = 0; j < 4; ++j) {
        int n = tn + j * 16 + l16;
        nok[j] = (n < N);
        Brow[j] = WT + (size_t)(nok[j] ? n : 0) * K + half16 * 16;
    }

    FragBF a0[4], b0[4], a1[4], b1[4];

    auto loadA = [&](FragBF (&a)[4], int k0) {
#pragma unroll
        for (int i = 0; i < 4; ++i) {
            const __bf16* p = Arow[i] + k0;
            a[i].u[0] = *(const uint4*)(p);
            a[i].u[1] = *(const uint4*)(p + 8);
        }
    };
    auto loadB = [&](FragBF (&b)[4], int k0) {
#pragma unroll
        for (int j = 0; j < 4; ++j) {
            if (nok[j]) {
                const __bf16* p = Brow[j] + k0;
                b[j].u[0] = *(const uint4*)(p);
                b[j].u[1] = *(const uint4*)(p + 8);
            } else {
                b[j].u[0] = make_uint4(0, 0, 0, 0);
                b[j].u[1] = make_uint4(0, 0, 0, 0);
            }
        }
    };
    auto mma = [&](FragBF (&a)[4], FragBF (&b)[4]) {
#pragma unroll
        for (int i = 0; i < 4; ++i)
#pragma unroll
            for (int j = 0; j < 4; ++j)
                acc[i][j] = wmma_bf16(a[i].v, b[j].v, acc[i][j]);
    };

    loadA(a0, 0);
    loadB(b0, 0);
    for (int k0 = 0; k0 < K; k0 += 64) {
        loadA(a1, k0 + 32);            // in flight under stage-0 WMMAs
        loadB(b1, k0 + 32);
        mma(a0, b0);
        if (k0 + 64 < K) {
            loadA(a0, k0 + 64);        // in flight under stage-1 WMMAs
            loadB(b0, k0 + 64);
        }
        mma(a1, b1);
        // pin interleave: 32 VMEM reads + 32 WMMAs per iteration as
        // 16 groups of (2 loads, 2 WMMAs)
#pragma unroll
        for (int g = 0; g < 16; ++g) {
            SGB(SG_VMREAD, 2);
            SGB(SG_WMMA, 2);
        }
    }

    // epilogue
#pragma unroll
    for (int j = 0; j < 4; ++j) {
        int n = tn + j * 16 + l16;
        if (n >= N) continue;
        float bs = bias ? bias[n] : 0.f;
        float div = 0.f;
        if (PENC)
            div = __expf(-(float)(n & ~1) * (9.210340371976184f / 512.f));
#pragma unroll
        for (int i = 0; i < 4; ++i) {
#pragma unroll
            for (int r = 0; r < 8; ++r) {
                int m = tm + i * 16 + r + half16 * 8;
                float v = acc[i][j][r] + bs;
                if (PENC) {
                    float s = (float)(m & (SEQ_ - 1));
                    v += (n & 1) ? __cosf(s * div) : __sinf(s * div);
                }
                if (RESD) v += resid[(size_t)m * N + n];
                if (GELU) v = 0.5f * v * (1.f + erff(v * 0.70710678118f));
                if (ELU1) v = (v > 0.f) ? (v + 1.f) : __expf(v);
                size_t off = (size_t)m * N + n;
                if (outF) outF[off] = v;
                if (outB) outB[off] = f2bf(v);
            }
        }
    }
}

// ---------------------------------------------------------------------------
// LayerNorm over last dim (512): one wave per row. fp32 in, fp32+bf16 out.
// ---------------------------------------------------------------------------
__global__ __launch_bounds__(256) void ln_k(
    const float* __restrict__ X, const float* __restrict__ g,
    const float* __restrict__ b, float* __restrict__ outF,
    __bf16* __restrict__ outB, int M)
{
    int lane = threadIdx.x & 31;
    int row  = blockIdx.x * 8 + (threadIdx.x >> 5);
    if (row >= M) return;
    const float* xr = X + (size_t)row * DMODEL;
    float xv[16];
    float s = 0.f;
#pragma unroll
    for (int t = 0; t < 16; ++t) { xv[t] = xr[lane + t * 32]; s += xv[t]; }
#pragma unroll
    for (int d = 16; d >= 1; d >>= 1) s += __shfl_xor(s, d, 32);
    float mu = s * (1.f / DMODEL);
    float q = 0.f;
#pragma unroll
    for (int t = 0; t < 16; ++t) { float d = xv[t] - mu; q += d * d; }
#pragma unroll
    for (int d = 16; d >= 1; d >>= 1) q += __shfl_xor(q, d, 32);
    float rs = rsqrtf(q * (1.f / DMODEL) + 1e-5f);
#pragma unroll
    for (int t = 0; t < 16; ++t) {
        int col = lane + t * 32;
        float v = (xv[t] - mu) * rs * g[col] + b[col];
        size_t off = (size_t)row * DMODEL + col;
        outF[off] = v;
        outB[off] = f2bf(v);
    }
}

// ---------------------------------------------------------------------------
// Chunked causal linear attention. One block (8 waves) per (b,h).
// q,k already have elu(x)+1 applied. Layout: [B,S,H*Dh] bf16.
// Dynamic LDS layout (bytes):
//   kT  bf16 [64][136]   @ 0       (17408)   k^T of current chunk
//   vT  bf16 [64][136]   @ 17408   (17408)   v^T of current chunk
//   sc  bf16 [128][136]  @ 34816   (34816)   masked scores (A-layout source)
//   ST  f32  [64][68]    @ 69632   (17408)   running state S^T[e][d]
//   zb  f32  [64]        @ 87040   (256)
//   den f32  [128]       @ 87296   (512)     total 87808 B
// ---------------------------------------------------------------------------
#define ATTN_LDS 87808

__global__ __launch_bounds__(256) void attn_k(
    const __bf16* __restrict__ q, const __bf16* __restrict__ k,
    const __bf16* __restrict__ v, __bf16* __restrict__ o)
{
    extern __shared__ __align__(16) char smem[];
    __bf16* kTp  = (__bf16*)(smem);
    __bf16* vTp  = (__bf16*)(smem + 17408);
    __bf16* scp  = (__bf16*)(smem + 34816);
    float*  STp  = (float*)(smem + 69632);
    float*  zbp  = (float*)(smem + 87040);
    float*  denp = (float*)(smem + 87296);

    const int bh = blockIdx.x;            // 0..31
    const int bb = bh >> 3, hh = bh & 7;
    const size_t base = ((size_t)bb * SEQ_) * DMODEL + hh * DHEAD;
    const __bf16* qp = q + base;
    const __bf16* kp = k + base;
    const __bf16* vp = v + base;
    __bf16* op = o + base;

    const int tid    = threadIdx.x;
    const int lane   = tid & 31;
    const int wave   = tid >> 5;
    const int l16    = lane & 15;
    const int half16 = lane >> 4;
    const int m0     = wave * 16;

    for (int i = tid; i < 64 * 68; i += 256) STp[i] = 0.f;
    if (tid < 64) zbp[tid] = 0.f;
    __syncthreads();

    for (int c = 0; c < NCHK; ++c) {
        const int sb = c * CHK;
        // ---- stage k^T, v^T of this chunk into LDS (b128 loads, b16 scatter)
        for (int seg = tid; seg < CHK * 8; seg += 256) {
            int srow = seg >> 3, d0 = (seg & 7) * 8;
            uint4 kv = *(const uint4*)(kp + (size_t)(sb + srow) * DMODEL + d0);
            uint4 vv = *(const uint4*)(vp + (size_t)(sb + srow) * DMODEL + d0);
            const __bf16* kh = (const __bf16*)&kv;
            const __bf16* vh = (const __bf16*)&vv;
#pragma unroll
            for (int t = 0; t < 8; ++t) {
                kTp[(d0 + t) * 136 + srow] = kh[t];
                vTp[(d0 + t) * 136 + srow] = vh[t];
            }
        }
        __syncthreads();

        // ---- scores = q_chunk[m0..m0+15] @ k_chunk^T  (16x128, K=64) ----
        v8f sa[8];
#pragma unroll
        for (int j = 0; j < 8; ++j) sa[j] = (v8f){0,0,0,0,0,0,0,0};
#pragma unroll
        for (int kk = 0; kk < DHEAD; kk += 32) {
            FragBF af;
            const __bf16* qr = qp + (size_t)(sb + m0 + l16) * DMODEL + kk + half16 * 16;
            af.u[0] = *(const uint4*)(qr);
            af.u[1] = *(const uint4*)(qr + 8);
            FragBF kb[8];
#pragma unroll
            for (int j = 0; j < 8; ++j) {
                const __bf16* kr = kp + (size_t)(sb + j * 16 + l16) * DMODEL + kk + half16 * 16;
                kb[j].u[0] = *(const uint4*)(kr);
                kb[j].u[1] = *(const uint4*)(kr + 8);
            }
#pragma unroll
            for (int j = 0; j < 8; ++j)
                sa[j] = wmma_bf16(af.v, kb[j].v, sa[j]);
            // pin: all 18 fragment loads issue ahead of the 8 WMMAs so the
            // per-WMMA waits become partial loadcnt waits, not 0.
            SGB(SG_VMREAD, 18);
            SGB(SG_WMMA, 8);
        }
        // ---- causal mask within chunk, store to LDS (bf16) ----
#pragma unroll
        for (int j = 0; j < 8; ++j) {
            int jj = j * 16 + l16;
#pragma unroll
            for (int r = 0; r < 8; ++r) {
                int i = m0 + r + half16 * 8;
                float val = (jj <= i) ? sa[j][r] : 0.f;
                scp[i * 136 + jj] = f2bf(val);
            }
        }
        __syncthreads();

        // ---- denominator: rowsum(scores) + q . z_old ----
        if (tid < CHK) {
            float s = 0.f;
            for (int j = 0; j < CHK; ++j) s += bf2f(scp[tid * 136 + j]);
            const __bf16* qr = qp + (size_t)(sb + tid) * DMODEL;
            float s2 = 0.f;
            for (int d = 0; d < DHEAD; ++d) s2 += bf2f(qr[d]) * zbp[d];
            denp[tid] = s + s2;
        }

        // ---- numerator = scores @ v_chunk (K=128) + q @ S_old (K=64) ----
        v8f na[4];
#pragma unroll
        for (int j = 0; j < 4; ++j) na[j] = (v8f){0,0,0,0,0,0,0,0};
#pragma unroll
        for (int kk = 0; kk < CHK; kk += 32) {
            FragBF af;
            const __bf16* sr = scp + (m0 + l16) * 136 + kk + half16 * 16;
            af.u[0] = *(const uint4*)(sr);
            af.u[1] = *(const uint4*)(sr + 8);
            FragBF vb[4];
#pragma unroll
            for (int j = 0; j < 4; ++j) {
                const __bf16* vr = vTp + (j * 16 + l16) * 136 + kk + half16 * 16;
                vb[j].u[0] = *(const uint4*)(vr);
                vb[j].u[1] = *(const uint4*)(vr + 8);
            }
#pragma unroll
            for (int j = 0; j < 4; ++j)
                na[j] = wmma_bf16(af.v, vb[j].v, na[j]);
            // pin: 10 LDS b128 reads ahead of the 4 WMMAs
            SGB(SG_DSREAD, 10);
            SGB(SG_WMMA, 4);
        }
#pragma unroll
        for (int kk = 0; kk < DHEAD; kk += 32) {
            FragBF af;
            const __bf16* qr = qp + (size_t)(sb + m0 + l16) * DMODEL + kk + half16 * 16;
            af.u[0] = *(const uint4*)(qr);
            af.u[1] = *(const uint4*)(qr + 8);
            FragBF sb_[4];
#pragma unroll
            for (int j = 0; j < 4; ++j) {
                const float* sr = STp + (j * 16 + l16) * 68 + kk + half16 * 16;
#pragma unroll
                for (int t = 0; t < 16; ++t) sb_[j].h[t] = f2bf(sr[t]);
            }
#pragma unroll
            for (int j = 0; j < 4; ++j)
                na[j] = wmma_bf16(af.v, sb_[j].v, na[j]);
        }
        __syncthreads();   // den visible; all reads of old ST/z complete

        // ---- out = num / (den + eps) ----
#pragma unroll
        for (int j = 0; j < 4; ++j) {
            int d = j * 16 + l16;
#pragma unroll
            for (int r = 0; r < 8; ++r) {
                int i = m0 + r + half16 * 8;
                float val = na[j][r] / (denp[i] + 1e-6f);
                op[(size_t)(sb + i) * DMODEL + d] = f2bf(val);
            }
        }

        // ---- state update: S^T += v^T @ k  (64x64, K=128); 2 tiles/wave ----
#pragma unroll
        for (int u = 0; u < 2; ++u) {
            int tt = wave * 2 + u;
            int et = tt >> 2, dt = tt & 3;
            v8f ca;
#pragma unroll
            for (int r = 0; r < 8; ++r)
                ca[r] = STp[(et * 16 + r + half16 * 8) * 68 + dt * 16 + l16];
#pragma unroll
            for (int kk = 0; kk < CHK; kk += 32) {
                FragBF af, bf_;
                const __bf16* vr = vTp + (et * 16 + l16) * 136 + kk + half16 * 16;
                af.u[0] = *(const uint4*)(vr);
                af.u[1] = *(const uint4*)(vr + 8);
                const __bf16* kr = kTp + (dt * 16 + l16) * 136 + kk + half16 * 16;
                bf_.u[0] = *(const uint4*)(kr);
                bf_.u[1] = *(const uint4*)(kr + 8);
                ca = wmma_bf16(af.v, bf_.v, ca);
            }
#pragma unroll
            for (int r = 0; r < 8; ++r)
                STp[(et * 16 + r + half16 * 8) * 68 + dt * 16 + l16] = ca[r];
        }
        // ---- z += colsum(k_chunk) ----
        if (tid < 64) {
            float s = 0.f;
            for (int j = 0; j < CHK; ++j) s += bf2f(kTp[tid * 136 + j]);
            zbp[tid] += s;
        }
        __syncthreads();
    }
}

// ---------------------------------------------------------------------------
// Host-side orchestration
// ---------------------------------------------------------------------------
static void launch_gemm(int flags, const __bf16* A, const __bf16* WT,
                        const float* bias, const float* resid,
                        float* oF, __bf16* oB, int M, int N, int K,
                        hipStream_t st)
{
    int tiles = (M >> 6) * ((N + 63) >> 6);
    int blocks = (tiles + 7) / 8;
    switch (flags) {
    case 0: gemm_k<0><<<blocks, 256, 0, st>>>(A, WT, bias, resid, oF, oB, M, N, K); break;
    case 1: gemm_k<1><<<blocks, 256, 0, st>>>(A, WT, bias, resid, oF, oB, M, N, K); break;
    case 2: gemm_k<2><<<blocks, 256, 0, st>>>(A, WT, bias, resid, oF, oB, M, N, K); break;
    case 4: gemm_k<4><<<blocks, 256, 0, st>>>(A, WT, bias, resid, oF, oB, M, N, K); break;
    case 8: gemm_k<8><<<blocks, 256, 0, st>>>(A, WT, bias, resid, oF, oB, M, N, K); break;
    }
}

static void launch_transpose(const float* W, __bf16* WT, int K, int N,
                             hipStream_t st)
{
    int blocks = (K >> 5) * (N >> 5);
    transposeW<<<blocks, 256, 0, st>>>(W, WT, K, N);
}

extern "C" void kernel_launch(void* const* d_in, const int* in_sizes, int n_in,
                              void* d_out, int out_size, void* d_ws, size_t ws_size,
                              hipStream_t stream)
{
    (void)in_sizes; (void)n_in; (void)out_size; (void)ws_size;
    const int M = MTOK;

    // ---- input pointers (setup_inputs dict order, nested dicts flattened) ----
    const int*   x    = (const int*)d_in[0];
    // d_in[1] = target (unused)
    const float* e0 = (const float*)d_in[2];
    const float* e1 = (const float*)d_in[3];
    const float* e2 = (const float*)d_in[4];
    const float* e3 = (const float*)d_in[5];
    const float* e4 = (const float*)d_in[6];
    const float* e5 = (const float*)d_in[7];
    const float* in_w = (const float*)d_in[8];
    const float* in_b = (const float*)d_in[9];
    // layers at 10 + 16*l ; proj_w at 202.. ; proj_b at 208..

    // ---- workspace carve ----
    size_t off = 0;
    auto alloc = [&](size_t bytes) {
        void* p = (char*)d_ws + off;
        off += (bytes + 255) & ~(size_t)255;
        return p;
    };
    float*  hF   = (float*) alloc((size_t)M * DMODEL * 4);
    __bf16* hB   = (__bf16*)alloc((size_t)M * DMODEL * 2);
    float*  tmpF = (float*) alloc((size_t)M * DMODEL * 4);
    __bf16* qB   = (__bf16*)alloc((size_t)M * DMODEL * 2);
    __bf16* kB   = (__bf16*)alloc((size_t)M * DMODEL * 2);
    __bf16* vB   = (__bf16*)alloc((size_t)M * DMODEL * 2);
    __bf16* aB   = (__bf16*)alloc((size_t)M * DMODEL * 2);
    __bf16* ffB  = (__bf16*)alloc((size_t)M * 2048 * 2);
    __bf16* embB = (__bf16*)alloc((size_t)M * EMBCAT * 2);
    __bf16* wtB  = (__bf16*)alloc((size_t)2048 * 512 * 2);

    // ---- embedding + input projection + positional encoding ----
    {
        int total = M * EMBCAT;
        embed_k<<<(total + 255) / 256, 256, 0, stream>>>(x, e0, e1, e2, e3, e4, e5, embB);
        launch_transpose(in_w, wtB, EMBCAT, DMODEL, stream);
        launch_gemm(4 /*POSENC*/, embB, wtB, in_b, nullptr, hF, hB,
                    M, DMODEL, EMBCAT, stream);
    }

    // ---- encoder layers ----
    for (int l = 0; l < 12; ++l) {
        const int base = 10 + 16 * l;
        const float* Wq  = (const float*)d_in[base + 0];
        const float* bq  = (const float*)d_in[base + 1];
        const float* Wk  = (const float*)d_in[base + 2];
        const float* bk  = (const float*)d_in[base + 3];
        const float* Wv  = (const float*)d_in[base + 4];
        const float* bv  = (const float*)d_in[base + 5];
        const float* Wo  = (const float*)d_in[base + 6];
        const float* bo  = (const float*)d_in[base + 7];
        const float* l1w = (const float*)d_in[base + 8];
        const float* l1b = (const float*)d_in[base + 9];
        const float* l2w = (const float*)d_in[base + 10];
        const float* l2b = (const float*)d_in[base + 11];
        const float* f1w = (const float*)d_in[base + 12];
        const float* f1b = (const float*)d_in[base + 13];
        const float* f2w = (const float*)d_in[base + 14];
        const float* f2b = (const float*)d_in[base + 15];

        launch_transpose(Wq, wtB, DMODEL, DMODEL, stream);
        launch_gemm(1 /*ELU1*/, hB, wtB, bq, nullptr, nullptr, qB, M, DMODEL, DMODEL, stream);
        launch_transpose(Wk, wtB, DMODEL, DMODEL, stream);
        launch_gemm(1 /*ELU1*/, hB, wtB, bk, nullptr, nullptr, kB, M, DMODEL, DMODEL, stream);
        launch_transpose(Wv, wtB, DMODEL, DMODEL, stream);
        launch_gemm(0, hB, wtB, bv, nullptr, nullptr, vB, M, DMODEL, DMODEL, stream);

        attn_k<<<MB_ * NHEAD, 256, ATTN_LDS, stream>>>(qB, kB, vB, aB);

        launch_transpose(Wo, wtB, DMODEL, DMODEL, stream);
        launch_gemm(8 /*RESID*/, aB, wtB, bo, hF, tmpF, nullptr, M, DMODEL, DMODEL, stream);
        ln_k<<<M / 8, 256, 0, stream>>>(tmpF, l1w, l1b, hF, hB, M);

        launch_transpose(f1w, wtB, DMODEL, 2048, stream);
        launch_gemm(2 /*GELU*/, hB, wtB, f1b, nullptr, nullptr, ffB, M, 2048, DMODEL, stream);
        launch_transpose(f2w, wtB, 2048, DMODEL, stream);
        launch_gemm(8 /*RESID*/, ffB, wtB, f2b, hF, tmpF, nullptr, M, DMODEL, 2048, stream);
        ln_k<<<M / 8, 256, 0, stream>>>(tmpF, l2w, l2b, hF, hB, M);
    }

    // ---- output projections (6 heads) ----
    const int NTOK[6] = {64, 256, 32, 512, 64, 64};
    float* outp = (float*)d_out;
    for (int i = 0; i < 6; ++i) {
        const float* pw = (const float*)d_in[202 + i];
        const float* pb = (const float*)d_in[208 + i];
        launch_transpose(pw, wtB, DMODEL, NTOK[i], stream);
        launch_gemm(0, hB, wtB, pb, nullptr, outp, nullptr, M, NTOK[i], DMODEL, stream);
        outp += (size_t)M * NTOK[i];
    }
}
